// HLRModel_64372969833040
// MI455X (gfx1250) — matrix operand order repaired and checked by reading, not compiled, not run on hardware
//
#include <hip/hip_runtime.h>
#include <hip/hip_bf16.h>

// ---------------------------------------------------------------------------
// Model constants (from the reference)
// ---------------------------------------------------------------------------
#define T_SEQ  1024
#define DMODEL 1024
#define NHEAD  16
#define HDIM   64
#define VOCAB  32000
#define FFDIM  4096
#define NLAYER 24
#define NGROUP 16
#define GDIM   64
#define NTRI   2016

typedef __attribute__((ext_vector_type(16))) __bf16 v16bf;
typedef __attribute__((ext_vector_type(8)))  __bf16 v8bf;
typedef __attribute__((ext_vector_type(8)))  float  v8f;
typedef __attribute__((ext_vector_type(4)))  unsigned v4u;
typedef __attribute__((ext_vector_type(8)))  unsigned v8u;

__device__ __forceinline__ __bf16 f2bf(float f) {
  unsigned u = __float_as_uint(f);
  u += 0x7FFFu + ((u >> 16) & 1u);          // round-to-nearest-even
  unsigned short h = (unsigned short)(u >> 16);
  __bf16 r;
  __builtin_memcpy(&r, &h, sizeof(r));
  return r;
}

__device__ __forceinline__ v8f vzero8() {
  v8f z;
#pragma unroll
  for (int i = 0; i < 8; ++i) z[i] = 0.0f;
  return z;
}

__device__ __forceinline__ v8f wmma_bf16(v16bf a, v16bf b, v8f c) {
  // D = A(16x32) * B(32x16) + C, fp32 accumulate
  return __builtin_amdgcn_wmma_f32_16x16x32_bf16(false, a, false, b,
                                                 (short)0, c, false, false);
}

// ---------------------------------------------------------------------------
// TDM issue: DMA a 32(k) x 128(n) bf16 tile of row-major B (ld = N elements)
// into LDS at byte offset `lds`.  D# layout per CDNA5 ISA
// §8.3/§8.4.  Group1 is invariant per GEMM; group0 carries lds/global addr.
// ---------------------------------------------------------------------------
__device__ __forceinline__ v8u tdm_group1(int N, int K) {
  v8u g1;
  g1[0] = 1u << 16;                                    // data_size = 1 (2 bytes)
  g1[1] = ((unsigned)N & 0xFFFFu) << 16;               // tensor_dim0[15:0]
  g1[2] = ((unsigned)N >> 16) |                        // tensor_dim0[31:16]
          (((unsigned)K & 0xFFFFu) << 16);             // tensor_dim1[15:0]
  g1[3] = ((unsigned)K >> 16) | (128u << 16);          // dim1 hi | tile_dim0=128
  g1[4] = 32u;                                         // tile_dim1=32, tile_dim2=0
  g1[5] = (unsigned)N;                                 // tensor_dim0_stride[31:0]
  g1[6] = 0u;                                          // stride0 hi | stride1 lo
  g1[7] = 0u;                                          // stride1 hi
  return g1;
}

__device__ __forceinline__ void tdm_issue(const __bf16* B, int N, int n0blk,
                                          int k0, unsigned lds, v8u g1) {
  unsigned long long ga =
      (unsigned long long)(size_t)(B + (size_t)k0 * N + n0blk);
  v4u g0;
  g0[0] = 1u;                                          // count=1, no gather
  g0[1] = lds;                                         // lds_addr (bytes)
  g0[2] = (unsigned)ga;                                // global_addr[31:0]
  g0[3] = (unsigned)((ga >> 32) & 0x01FFFFFFu)         // global_addr[56:32]
          | (2u << 30);                                // type = 2 ("image")
  asm volatile("tensor_load_to_lds %0, %1" :: "s"(g0), "s"(g1) : "memory");
}

// ---------------------------------------------------------------------------
// fp32 -> bf16 elementwise convert
// ---------------------------------------------------------------------------
__global__ __launch_bounds__(256) void k_cvt(const float* __restrict__ a,
                                             __bf16* __restrict__ b, int n) {
  int i = blockIdx.x * 256 + threadIdx.x;
  if (i < n) b[i] = f2bf(a[i]);
}

// ---------------------------------------------------------------------------
// lm_head (V,D) fp32  ->  lmT (D,V) bf16  (LDS-tiled transpose)
// ---------------------------------------------------------------------------
__global__ __launch_bounds__(256) void k_tlm(const float* __restrict__ lm,
                                             __bf16* __restrict__ lmT) {
  __shared__ __bf16 tile[32][33];
  const int bv = blockIdx.x * 32;   // vocab tile
  const int bd = blockIdx.y * 32;   // dim tile
  const int tx = threadIdx.x & 31;
  const int ty = threadIdx.x >> 5;  // 8 rows per pass
#pragma unroll
  for (int i = 0; i < 32; i += 8)
    tile[ty + i][tx] = f2bf(lm[(size_t)(bv + ty + i) * DMODEL + bd + tx]);
  __syncthreads();
#pragma unroll
  for (int i = 0; i < 32; i += 8)
    lmT[(size_t)(bd + ty + i) * VOCAB + bv + tx] = tile[tx][ty + i];
}

// ---------------------------------------------------------------------------
// Cayley: R = (I - A/2)^-1 (I + A/2) per (layer, group); Gauss-Jordan in LDS.
// One block of 64 threads per (l,g); thread r owns row r.
// ---------------------------------------------------------------------------
#define LDW 130
__global__ __launch_bounds__(64) void k_cayley(const float* __restrict__ su,
                                               __bf16* __restrict__ Rb) {
  __shared__ float sh[64 * LDW];
  const int lg = blockIdx.x;          // l*NGROUP + g
  const int r  = threadIdx.x;
  const float* s = su + (size_t)lg * NTRI;
  for (int c = 0; c < 64; ++c) {
    float a;
    if (r < c)      { int idx = r * 63 - r * (r - 1) / 2 + (c - r - 1); a =  s[idx]; }
    else if (r > c) { int idx = c * 63 - c * (c - 1) / 2 + (r - c - 1); a = -s[idx]; }
    else a = 0.0f;
    float diag = (r == c) ? 1.0f : 0.0f;
    sh[r * LDW + c]      = diag - 0.5f * a;   // M = I - A/2
    sh[r * LDW + 64 + c] = diag + 0.5f * a;   // N = I + A/2
  }
  __syncthreads();
  for (int p = 0; p < 64; ++p) {
    float piv = sh[p * LDW + p];
    float f   = sh[r * LDW + p] / piv;
    __syncthreads();
    if (r != p)
      for (int c = p; c < 128; ++c) sh[r * LDW + c] -= f * sh[p * LDW + c];
    __syncthreads();
  }
  float inv = 1.0f / sh[r * LDW + r];
  for (int j = 0; j < 64; ++j)
    Rb[((size_t)lg * 64 + r) * 64 + j] = f2bf(sh[r * LDW + 64 + j] * inv);
}

// ---------------------------------------------------------------------------
// Embedding gather: x[t,:] = embed_w[tokens[t],:]
// ---------------------------------------------------------------------------
__global__ __launch_bounds__(256) void k_embed(const int* __restrict__ tok,
                                               const float* __restrict__ emb,
                                               float* __restrict__ x) {
  int t = blockIdx.x;
  int row = tok[t];
  for (int d = threadIdx.x; d < DMODEL; d += 256)
    x[(size_t)t * DMODEL + d] = emb[(size_t)row * DMODEL + d];
}

// ---------------------------------------------------------------------------
// RMSNorm (+ optional affine modulation) -> bf16.  One block per row.
// ---------------------------------------------------------------------------
__global__ __launch_bounds__(256) void k_norm(const float* __restrict__ x,
                                              const float* __restrict__ w,
                                              const float* __restrict__ gamma,
                                              const float* __restrict__ beta,
                                              __bf16* __restrict__ out) {
  __shared__ float red[8];
  const int t = blockIdx.x;
  const int lane = threadIdx.x & 31;
  const int wave = threadIdx.x >> 5;
  float ss = 0.0f;
  for (int d = threadIdx.x; d < DMODEL; d += 256) {
    float v = x[(size_t)t * DMODEL + d];
    ss += v * v;
  }
#pragma unroll
  for (int m = 1; m < 32; m <<= 1) ss += __shfl_xor(ss, m);
  if (lane == 0) red[wave] = ss;
  __syncthreads();
  float tot = 0.0f;
#pragma unroll
  for (int i = 0; i < 8; ++i) tot += red[i];
  const float rs = rsqrtf(tot * (1.0f / DMODEL) + 1e-6f);
  for (int d = threadIdx.x; d < DMODEL; d += 256) {
    float h = x[(size_t)t * DMODEL + d] * rs * w[d];
    if (gamma) h = h * gamma[d] + beta[d];
    out[(size_t)t * DMODEL + d] = f2bf(h);
  }
}

// ---------------------------------------------------------------------------
// bf16 WMMA GEMM with TDM-staged, double-buffered B tiles in LDS:
//   C(MxN,f32) = A(MxK,bf16,rm) @ B(KxN,bf16,rm)
// Block = 256 threads = 8 waves; wave -> 16x64 output; block -> 64x128.
// Wave 0 DMAs the 32x128 B tile for step s+1 while all 8 waves compute step s
// from LDS; synchronization via s_wait_tensorcnt + workgroup barrier.
// ---------------------------------------------------------------------------
__global__ __launch_bounds__(256) void k_gemm(const __bf16* __restrict__ A,
                                              const __bf16* __restrict__ B,
                                              float* __restrict__ C,
                                              int N, int K) {
  __shared__ __align__(128) __bf16 Bsm[2][32 * 128];
  const int wave = threadIdx.x >> 5;
  const int lane = threadIdx.x & 31;
  const int half = lane >> 4;
  const int lml  = lane & 15;
  const int m0 = blockIdx.y * 64 + (wave & 3) * 16;
  const int n0blk = blockIdx.x * 128;
  const int wc = wave >> 2;                 // wave column (0..1) -> 64 cols

  v8f acc[4];
#pragma unroll
  for (int t = 0; t < 4; ++t) acc[t] = vzero8();

  const __bf16* ap = A + (size_t)(m0 + lml) * K + half * 8;
  const unsigned lds0 = (unsigned)(size_t)(&Bsm[0][0]);
  const unsigned lds1 = (unsigned)(size_t)(&Bsm[1][0]);
  const v8u g1 = tdm_group1(N, K);

  const int nsteps = K / 32;
  if (threadIdx.x < 32)                     // wave 0 drives the TDM
    tdm_issue(B, N, n0blk, 0, lds0, g1);

  for (int s = 0; s < nsteps; ++s) {
    const int k0 = s * 32;
    if (threadIdx.x < 32) {
      if (s + 1 < nsteps) {
        tdm_issue(B, N, n0blk, k0 + 32, ((s + 1) & 1) ? lds1 : lds0, g1);
        __builtin_amdgcn_s_wait_tensorcnt(1);   // tile s complete (in-order)
      } else {
        __builtin_amdgcn_s_wait_tensorcnt(0);
      }
    }
    __syncthreads();

    v8bf alo = *(const v8bf*)(ap + k0);         // K = k0 + half*8 + {0..7}
    v8bf ahi = *(const v8bf*)(ap + k0 + 16);    // K = k0 + half*8 + {16..23}
    v16bf a;
#pragma unroll
    for (int i = 0; i < 8; ++i) { a[i] = alo[i]; a[8 + i] = ahi[i]; }
    if (k0 + 32 < K) __builtin_prefetch((const void*)(ap + k0 + 32), 0, 1);

    const __bf16* bsm = &Bsm[s & 1][0] + half * 8 * 128 + wc * 64 + lml;  // col N = lml, K = half*8 + {0..7, 16..23}
#pragma unroll
    for (int t = 0; t < 4; ++t) {
      v16bf b;
#pragma unroll
      for (int i = 0; i < 8; ++i) { b[i] = bsm[i * 128 + t * 16]; b[8 + i] = bsm[(16 + i) * 128 + t * 16]; }
      acc[t] = wmma_bf16(a, b, acc[t]);
    }
    __syncthreads();
  }
#pragma unroll
  for (int t = 0; t < 4; ++t)
#pragma unroll
    for (int v = 0; v < 8; ++v)
      C[(size_t)(m0 + v + 8 * half) * N + n0blk + wc * 64 + t * 16 + lml] =
          acc[t][v];
}

// ---------------------------------------------------------------------------
// qkv(T,3D) fp32 -> Qb(H,T,64) bf16, Ktb(H,64,T) bf16, Vb(H,T,64) bf16
// ---------------------------------------------------------------------------
__global__ __launch_bounds__(256) void k_split(const float* __restrict__ qkv,
                                               __bf16* __restrict__ Qb,
                                               __bf16* __restrict__ Ktb,
                                               __bf16* __restrict__ Vb) {
  int i = blockIdx.x * 256 + threadIdx.x;     // over T*DMODEL
  int t = i / DMODEL, hd = i % DMODEL;
  int h = hd / HDIM, d = hd % HDIM;
  const float* q = qkv + (size_t)t * (3 * DMODEL);
  Qb[((size_t)h * T_SEQ + t) * HDIM + d]   = f2bf(q[hd]);
  Ktb[((size_t)h * HDIM + d) * T_SEQ + t]  = f2bf(q[DMODEL + hd]);
  Vb[((size_t)h * T_SEQ + t) * HDIM + d]   = f2bf(q[2 * DMODEL + hd]);
}

// ---------------------------------------------------------------------------
// Fused causal flash attention. One wave per (16-query tile, head).
// S-tile = Q(16x64) @ Kt(64x16) via 2 WMMAs; online softmax; P through LDS
// back into A-fragment layout; O += P(16x32) @ V(32x64) via 4 WMMAs.
// ---------------------------------------------------------------------------
__global__ __launch_bounds__(32) void k_attn(const __bf16* __restrict__ Qb,
                                             const __bf16* __restrict__ Ktb,
                                             const __bf16* __restrict__ Vb,
                                             __bf16* __restrict__ Ob) {
  __shared__ __align__(32) __bf16 Pl[16 * 32];
  const int lane = threadIdx.x;
  const int half = lane >> 4;
  const int lml  = lane & 15;
  const int q0 = blockIdx.x * 16;
  const int h  = blockIdx.y;
  const __bf16* Q  = Qb  + (size_t)h * T_SEQ * HDIM;
  const __bf16* Kt = Ktb + (size_t)h * HDIM * T_SEQ;
  const __bf16* V  = Vb  + (size_t)h * T_SEQ * HDIM;

  v16bf aQ[2];
#pragma unroll
  for (int c = 0; c < 2; ++c) {
    const __bf16* qp = Q + (size_t)(q0 + lml) * HDIM + c * 32 + half * 8;
#pragma unroll
    for (int i = 0; i < 8; ++i) { aQ[c][i] = qp[i]; aQ[c][8 + i] = qp[16 + i]; }
  }

  v8f o[4];
#pragma unroll
  for (int t = 0; t < 4; ++t) o[t] = vzero8();
  float mrun[8], lrun[8];
#pragma unroll
  for (int v = 0; v < 8; ++v) { mrun[v] = -1e30f; lrun[v] = 0.0f; }

  const int limit = q0 + 16;             // causal: keys 0..q inclusive
  const float scale = 0.125f;            // 1/sqrt(64)

  for (int k0 = 0; k0 < limit; k0 += 32) {
    float s[2][8];
#pragma unroll
    for (int sub = 0; sub < 2; ++sub) {
      const int kk = k0 + sub * 16;
      if (kk < limit) {
        v8f c = vzero8();
#pragma unroll
        for (int c2 = 0; c2 < 2; ++c2) {
          const __bf16* kp = Kt + (size_t)(c2 * 32 + half * 8) * T_SEQ + kk + lml;
          v16bf b;
#pragma unroll
          for (int i = 0; i < 8; ++i) { b[i] = kp[(size_t)i * T_SEQ]; b[8 + i] = kp[(size_t)(16 + i) * T_SEQ]; }
          c = wmma_bf16(aQ[c2], b, c);
        }
#pragma unroll
        for (int v = 0; v < 8; ++v) {
          int key = kk + lml;                    // N index
          int qy  = q0 + v + 8 * half;           // M index
          s[sub][v] = (key <= qy) ? c[v] * scale : -1e30f;
        }
      } else {
#pragma unroll
        for (int v = 0; v < 8; ++v) s[sub][v] = -1e30f;
      }
    }
    // online softmax across keys (lanes within 16-group, plus both subtiles)
#pragma unroll
    for (int v = 0; v < 8; ++v) {
      float cm = fmaxf(s[0][v], s[1][v]);
#pragma unroll
      for (int m = 1; m < 16; m <<= 1) cm = fmaxf(cm, __shfl_xor(cm, m));
      float nm = fmaxf(mrun[v], cm);
      float corr = __expf(mrun[v] - nm);
      mrun[v] = nm;
      float p0 = __expf(s[0][v] - nm);
      float p1 = __expf(s[1][v] - nm);
      float ps = p0 + p1;
#pragma unroll
      for (int m = 1; m < 16; m <<= 1) ps += __shfl_xor(ps, m);
      lrun[v] = lrun[v] * corr + ps;
#pragma unroll
      for (int t = 0; t < 4; ++t) o[t][v] *= corr;   // same M layout as S
      Pl[(v + 8 * half) * 32 + lml]      = f2bf(p0); // P[q][k_local]
      Pl[(v + 8 * half) * 32 + 16 + lml] = f2bf(p1);
    }
    __syncthreads();
    v16bf aP;
    const __bf16* pp = Pl + lml * 32 + half * 8;
#pragma unroll
    for (int i = 0; i < 8; ++i) { aP[i] = pp[i]; aP[8 + i] = pp[16 + i]; }
#pragma unroll
    for (int t = 0; t < 4; ++t) {
      const __bf16* vp = V + (size_t)(k0 + half * 8) * HDIM + t * 16 + lml;
      v16bf bV;
#pragma unroll
      for (int i = 0; i < 8; ++i) { bV[i] = vp[i * HDIM]; bV[8 + i] = vp[(16 + i) * HDIM]; }
      o[t] = wmma_bf16(aP, bV, o[t]);
    }
    __syncthreads();
  }
#pragma unroll
  for (int t = 0; t < 4; ++t)
#pragma unroll
    for (int v = 0; v < 8; ++v) {
      float val = o[t][v] / lrun[v];
      Ob[(size_t)(q0 + v + 8 * half) * DMODEL + h * HDIM + t * 16 + lml] = f2bf(val);
    }
}

// ---------------------------------------------------------------------------
// Elementwise kernels
// ---------------------------------------------------------------------------
__global__ __launch_bounds__(256) void k_add(const float* __restrict__ a,
                                             const float* __restrict__ b,
                                             float* __restrict__ c, int n) {
  int i = blockIdx.x * 256 + threadIdx.x;
  if (i < n) c[i] = a[i] + b[i];
}

__global__ __launch_bounds__(256) void k_gelu(const float* __restrict__ a,
                                              __bf16* __restrict__ b, int n) {
  int i = blockIdx.x * 256 + threadIdx.x;
  if (i < n) {
    float x = a[i];
    float t = tanhf(0.7978845608028654f * (x + 0.044715f * x * x * x));
    b[i] = f2bf(0.5f * x * (1.0f + t));
  }
}

// x' = x + ((y + ff2) - x) * iter_scale[l]
__global__ __launch_bounds__(256) void k_blend(const float* __restrict__ x,
                                               const float* __restrict__ y,
                                               const float* __restrict__ ff2,
                                               const float* __restrict__ its,
                                               int l, float* __restrict__ xt,
                                               int n) {
  int i = blockIdx.x * 256 + threadIdx.x;
  if (i < n) {
    float s = its[l];
    float blk = y[i] + ff2[i];
    xt[i] = x[i] + (blk - x[i]) * s;
  }
}

// ---------------------------------------------------------------------------
// Group rotation via WMMA: x_out[t, g*64+j] = sum_i x_in[t, g*64+i] * R[l,g,i,j]
// One wave per (16-row tile, group).
// ---------------------------------------------------------------------------
__global__ __launch_bounds__(32) void k_rot(const float* __restrict__ xin,
                                            const __bf16* __restrict__ Rb,
                                            float* __restrict__ xout, int layer) {
  const int lane = threadIdx.x;
  const int half = lane >> 4;
  const int lml  = lane & 15;
  const int t0  = blockIdx.x * 16;
  const int grp = blockIdx.y;
  const __bf16* R = Rb + ((size_t)layer * NGROUP + grp) * GDIM * GDIM;
  v8f acc[4];
#pragma unroll
  for (int t = 0; t < 4; ++t) acc[t] = vzero8();

#pragma unroll
  for (int c = 0; c < 2; ++c) {
    const int k0 = c * 32;
    const float* ap = xin + (size_t)(t0 + lml) * DMODEL + grp * GDIM + k0 + half * 8;
    v16bf a;
#pragma unroll
    for (int i = 0; i < 8; ++i) { a[i] = f2bf(ap[i]); a[8 + i] = f2bf(ap[16 + i]); }
    const __bf16* bp = R + (size_t)(k0 + half * 8) * GDIM + lml;
#pragma unroll
    for (int t = 0; t < 4; ++t) {
      v16bf b;
#pragma unroll
      for (int i = 0; i < 8; ++i) { b[i] = bp[i * GDIM + t * 16]; b[8 + i] = bp[(16 + i) * GDIM + t * 16]; }
      acc[t] = wmma_bf16(a, b, acc[t]);
    }
  }
#pragma unroll
  for (int t = 0; t < 4; ++t)
#pragma unroll
    for (int v = 0; v < 8; ++v)
      xout[(size_t)(t0 + v + 8 * half) * DMODEL + grp * GDIM + t * 16 + lml] = acc[t][v];
}

// ---------------------------------------------------------------------------
// Host-side orchestration
// ---------------------------------------------------------------------------
extern "C" void kernel_launch(void* const* d_in, const int* in_sizes, int n_in,
                              void* d_out, int out_size, void* d_ws, size_t ws_size,
                              hipStream_t stream) {
  (void)in_sizes; (void)n_in; (void)out_size; (void)ws_size;
  const int*   tokens      = (const int*)d_in[0];
  const float* embed_w     = (const float*)d_in[1];
  const float* lm_head_w   = (const float*)d_in[2];
  const float* norm_w      = (const float*)d_in[3];
  const float* layer_gamma = (const float*)d_in[4];
  const float* layer_beta  = (const float*)d_in[5];
  const float* iter_scale  = (const float*)d_in[6];
  const float* skew_upper  = (const float*)d_in[7];
  const float* ln1_w       = (const float*)d_in[8];
  const float* ln2_w       = (const float*)d_in[9];
  const float* wqkv        = (const float*)d_in[10];
  const float* wo          = (const float*)d_in[11];
  const float* w1          = (const float*)d_in[12];
  const float* w2          = (const float*)d_in[13];
  float* out = (float*)d_out;

  // ---- workspace carve-up ----
  char* wsp = (char*)d_ws;
  size_t off = 0;
  auto alloc = [&](size_t bytes) -> void* {
    void* p = (void*)(wsp + off);
    off += (bytes + 255) & ~(size_t)255;
    return p;
  };
  __bf16* wqkv_b = (__bf16*)alloc((size_t)DMODEL * 3 * DMODEL * 2);
  __bf16* wo_b   = (__bf16*)alloc((size_t)DMODEL * DMODEL * 2);
  __bf16* w1_b   = (__bf16*)alloc((size_t)DMODEL * FFDIM * 2);
  __bf16* w2_b   = (__bf16*)alloc((size_t)FFDIM * DMODEL * 2);
  __bf16* lmT    = (__bf16*)alloc((size_t)DMODEL * VOCAB * 2);
  __bf16* Rb     = (__bf16*)alloc((size_t)NLAYER * NGROUP * GDIM * GDIM * 2);
  float*  x      = (float*)alloc((size_t)T_SEQ * DMODEL * 4);
  float*  y      = (float*)alloc((size_t)T_SEQ * DMODEL * 4);
  float*  xt     = (float*)alloc((size_t)T_SEQ * DMODEL * 4);
  __bf16* hbuf   = (__bf16*)alloc((size_t)T_SEQ * DMODEL * 2);
  float*  qkvf   = (float*)alloc((size_t)T_SEQ * 3 * DMODEL * 4);
  __bf16* Qb     = (__bf16*)alloc((size_t)NHEAD * T_SEQ * HDIM * 2);
  __bf16* Ktb    = (__bf16*)alloc((size_t)NHEAD * HDIM * T_SEQ * 2);
  __bf16* Vb     = (__bf16*)alloc((size_t)NHEAD * T_SEQ * HDIM * 2);
  __bf16* attnb  = (__bf16*)alloc((size_t)T_SEQ * DMODEL * 2);
  float*  projf  = (float*)alloc((size_t)T_SEQ * DMODEL * 4);
  float*  ff1f   = (float*)alloc((size_t)T_SEQ * FFDIM * 4);
  __bf16* gb     = (__bf16*)alloc((size_t)T_SEQ * FFDIM * 2);
  float*  ff2f   = (float*)alloc((size_t)T_SEQ * DMODEL * 4);
  __bf16* xn     = (__bf16*)alloc((size_t)T_SEQ * DMODEL * 2);

  const int TD = T_SEQ * DMODEL;

  // ---- one-time (per call) weight prep ----
  k_cvt<<<(DMODEL * 3 * DMODEL) / 256, 256, 0, stream>>>(wqkv, wqkv_b, DMODEL * 3 * DMODEL);
  k_cvt<<<(DMODEL * DMODEL) / 256, 256, 0, stream>>>(wo, wo_b, DMODEL * DMODEL);
  k_cvt<<<(DMODEL * FFDIM) / 256, 256, 0, stream>>>(w1, w1_b, DMODEL * FFDIM);
  k_cvt<<<(FFDIM * DMODEL) / 256, 256, 0, stream>>>(w2, w2_b, FFDIM * DMODEL);
  k_tlm<<<dim3(VOCAB / 32, DMODEL / 32), 256, 0, stream>>>(lm_head_w, lmT);
  k_cayley<<<NLAYER * NGROUP, 64, 0, stream>>>(skew_upper, Rb);
  k_embed<<<T_SEQ, 256, 0, stream>>>(tokens, embed_w, x);

  // ---- layer loop ----
  for (int l = 0; l < NLAYER; ++l) {
    // h = rmsnorm(x, ln1)*gamma + beta  (bf16)
    k_norm<<<T_SEQ, 256, 0, stream>>>(x, ln1_w,
                                      layer_gamma + (size_t)l * DMODEL,
                                      layer_beta + (size_t)l * DMODEL, hbuf);
    // qkv = h @ wqkv
    k_gemm<<<dim3(3 * DMODEL / 128, T_SEQ / 64), 256, 0, stream>>>(hbuf, wqkv_b, qkvf,
                                                                   3 * DMODEL, DMODEL);
    k_split<<<TD / 256, 256, 0, stream>>>(qkvf, Qb, Ktb, Vb);
    k_attn<<<dim3(T_SEQ / 16, NHEAD), 32, 0, stream>>>(Qb, Ktb, Vb, attnb);
    // proj = attn @ wo ; y = x + proj
    k_gemm<<<dim3(DMODEL / 128, T_SEQ / 64), 256, 0, stream>>>(attnb, wo_b, projf,
                                                               DMODEL, DMODEL);
    k_add<<<TD / 256, 256, 0, stream>>>(x, projf, y, TD);
    // ffn on y
    k_norm<<<T_SEQ, 256, 0, stream>>>(y, ln2_w, nullptr, nullptr, hbuf);
    k_gemm<<<dim3(FFDIM / 128, T_SEQ / 64), 256, 0, stream>>>(hbuf, w1_b, ff1f,
                                                              FFDIM, DMODEL);
    k_gelu<<<(T_SEQ * FFDIM) / 256, 256, 0, stream>>>(ff1f, gb, T_SEQ * FFDIM);
    k_gemm<<<dim3(DMODEL / 128, T_SEQ / 64), 256, 0, stream>>>(gb, w2_b, ff2f,
                                                               DMODEL, FFDIM);
    // blend + group rotation
    k_blend<<<TD / 256, 256, 0, stream>>>(x, y, ff2f, iter_scale, l, xt, TD);
    k_rot<<<dim3(T_SEQ / 16, NGROUP), 32, 0, stream>>>(xt, Rb, x, l);
  }

  // ---- final norm + logits ----
  k_norm<<<T_SEQ, 256, 0, stream>>>(x, norm_w, nullptr, nullptr, xn);
  k_gemm<<<dim3(VOCAB / 128, T_SEQ / 64), 256, 0, stream>>>(xn, lmT, out, VOCAB, DMODEL);
}
